// PCT_Seg_57732950393335
// MI455X (gfx1250) — matrix operand order repaired and checked by reading, not compiled, not run on hardware
//
#include <hip/hip_runtime.h>
#include <hip/hip_bf16.h>

// ---------------------------------------------------------------------------
// CDNA5 (gfx1250) WMMA types / helpers
// ---------------------------------------------------------------------------
typedef __attribute__((ext_vector_type(16))) __bf16 v16bf;
typedef __attribute__((ext_vector_type(8)))  __bf16 v8bf;
typedef __attribute__((ext_vector_type(8)))  float  v8f;

__device__ __forceinline__ v8f wmma_bf16(v16bf a, v16bf b, v8f c) {
  // D = A(16x32 bf16) * B(32x16 bf16) + C(16x16 f32)
  return __builtin_amdgcn_wmma_f32_16x16x32_bf16(
      false, a, false, b, (short)0, c, false, false);
}

// ---- A fragment loaders (16-bit A 16x32, contiguous K per lane half) ----------------------
// lanes 0-15: row M=lane, halves 0..15 -> K 0..15
// lanes 16-31: same rows, K 16..31
// Fast path: fully in-bounds, row-major, ldA % 4 == 0  ->  4x global_load_b128
__device__ __forceinline__ v16bf load_A_fast(const float* __restrict__ W,
                                             long ld, int m0, int kb) {
  int lane = threadIdx.x & 31;
  int hi   = lane >> 4;
  const float* base = W + (long)(m0 + (lane & 15)) * ld + kb + (hi ? 16 : 0);
  float4 a0 = *(const float4*)(base);
  float4 a1 = *(const float4*)(base + 4);
  float4 a2 = *(const float4*)(base + 8);
  float4 a3 = *(const float4*)(base + 12);
  v16bf a;
  a[0] = (__bf16)a0.x;  a[1] = (__bf16)a0.y;  a[2]  = (__bf16)a0.z;  a[3]  = (__bf16)a0.w;
  a[4] = (__bf16)a1.x;  a[5] = (__bf16)a1.y;  a[6]  = (__bf16)a1.z;  a[7]  = (__bf16)a1.w;
  a[8] = (__bf16)a2.x;  a[9] = (__bf16)a2.y;  a[10] = (__bf16)a2.z;  a[11] = (__bf16)a2.w;
  a[12] = (__bf16)a3.x; a[13] = (__bf16)a3.y; a[14] = (__bf16)a3.z;  a[15] = (__bf16)a3.w;
  return a;
}

// Slow path: per-element predication / transposed A (small edge GEMMs only)
__device__ __forceinline__ v16bf load_A_slow(const float* __restrict__ W, long ld,
                                             int m0, int kb, int M, int K, int transA) {
  int lane = threadIdx.x & 31;
  int hi   = lane >> 4;
  int m    = m0 + (lane & 15);
  v16bf a;
#pragma unroll
  for (int h = 0; h < 16; ++h) {
    int k = kb + h + (hi ? 16 : 0);
    float v = 0.f;
    if (m < M && k < K)
      v = transA ? W[(long)k * ld + m] : W[(long)m * ld + k];
    a[h] = (__bf16)v;
  }
  return a;
}

// ---------------------------------------------------------------------------
// Generic WMMA GEMM:  Y[b] = act( bn( W[b] * X[b] + bias + bias2[b] ) )
// Block tile M128 x N64, 8 waves; wave owns 16 rows, 4 WMMAs per A fragment.
// X tile staged TRANSPOSED in LDS ([n][k] bf16) so each B fragment is one
// contiguous 32B run -> 2x ds_load_b128 (B 32x16: lane=col, 16 consec K, same K order as A).
// Double-buffered staging: tile i+1 streams from HBM while tile i feeds the
// WMMAs; single barrier per K-step.
// ---------------------------------------------------------------------------
__global__ __launch_bounds__(256)
void k_gemm(const float* __restrict__ W, long wbs, long ldA, int transA,
            const float* __restrict__ X, long xbs,
            float* __restrict__ Y, long ybs,
            int M, int K, int N,
            const float* __restrict__ bias, const float* __restrict__ bias2,
            const float* __restrict__ g,  const float* __restrict__ bt,
            const float* __restrict__ mn, const float* __restrict__ vr,
            int act) {
  __shared__ __bf16 sBT[2][64 * 32];  // ping-pong [n][k] tiles, 8 KB
  int b    = blockIdx.z;
  int n0   = blockIdx.x * 64;
  int wave = threadIdx.x >> 5;
  int m0   = blockIdx.y * 128 + wave * 16;
  const float* Wb = W + (long)b * wbs;
  const float* Xb = X + (long)b * xbs;
  int lane = threadIdx.x & 31;
  int col  = lane & 15, hi = lane >> 4;

  v8f acc[4] = {};
  int tk = threadIdx.x >> 4;         // 0..15 (k row)
  int tn = (threadIdx.x & 15) * 4;   // 0..60 (n quad)

  auto stage = [&](int kb, __bf16* dst) {
    if ((kb + 32 <= K) && (n0 + 64 <= N)) {     // uniform fast stage
#pragma unroll
      for (int kk = 0; kk < 32; kk += 16) {
        float4 v = *(const float4*)&Xb[(long)(kb + tk + kk) * N + n0 + tn];
        dst[(tn + 0) * 32 + tk + kk] = (__bf16)v.x;
        dst[(tn + 1) * 32 + tk + kk] = (__bf16)v.y;
        dst[(tn + 2) * 32 + tk + kk] = (__bf16)v.z;
        dst[(tn + 3) * 32 + tk + kk] = (__bf16)v.w;
      }
    } else {
      for (int i = threadIdx.x; i < 2048; i += 256) {
        int c = i & 63, kr = i >> 6;
        int k = kb + kr, n = n0 + c;
        float v = (k < K && n < N) ? Xb[(long)k * N + n] : 0.f;
        dst[c * 32 + kr] = (__bf16)v;
      }
    }
  };

  stage(0, sBT[0]);
  __syncthreads();

  for (int kb = 0, it = 0; kb < K; kb += 32, ++it) {
    const __bf16* cur = sBT[it & 1];
    if (kb + 32 < K)                     // overlap next-tile fetch with WMMAs
      stage(kb + 32, sBT[(it + 1) & 1]);

    v16bf af = (!transA && (m0 + 16 <= M) && (kb + 32 <= K))
                   ? load_A_fast(Wb, ldA, m0, kb)
                   : load_A_slow(Wb, ldA, m0, kb, M, K, transA);
    if (kb + 32 < K && m0 < M)   // stream next weight block (global_prefetch_b8)
      __builtin_prefetch(transA ? &Wb[(long)(kb + 32) * ldA + m0]
                                : &Wb[(long)m0 * ldA + kb + 32], 0, 1);
#pragma unroll
    for (int nt = 0; nt < 4; ++nt) {
      const v16bf* bp =
          (const v16bf*)&cur[((nt * 16 + col) << 5) + (hi ? 16 : 0)];
      acc[nt] = wmma_bf16(af, *bp, acc[nt]);
    }
    __syncthreads();   // next tile staged AND this tile fully consumed
  }

#pragma unroll
  for (int nt = 0; nt < 4; ++nt) {
    int n = n0 + nt * 16 + col;
#pragma unroll
    for (int r = 0; r < 8; ++r) {
      int m = m0 + r + (hi ? 8 : 0);
      if (m < M && n < N) {
        float v = acc[nt][r];
        if (bias)  v += bias[m];
        if (bias2) v += bias2[(long)b * M + m];
        if (g) {
          float sc = g[m] * rsqrtf(vr[m] + 1e-5f);
          v = (v - mn[m]) * sc + bt[m];
        }
        if (act == 1)      v = v > 0.f ? v : 0.f;
        else if (act == 2) v = v > 0.f ? v : 0.2f * v;
        Y[(long)b * ybs + (long)m * N + n] = v;
      }
    }
  }
}

// ---------------------------------------------------------------------------
// Fused local_op: gather group (grouped_norm || rep) -> GEMM1(bn,relu)
//                 -> GEMM2(bn,relu) -> max over 32 samples.  All in LDS.
// G/H1 stored TRANSPOSED ([sample][chan]) for contiguous B fragments; A always
// in-bounds -> b128 loads; one A fragment feeds both N-tiles (2 WMMAs).
// ci, CO are multiples of 32; CO/16 is a multiple of 8 (256/512 channels).
// ---------------------------------------------------------------------------
__global__ __launch_bounds__(256)
void k_local_op(const float* __restrict__ F, int D, int Np,
                const int* __restrict__ knn, const int* __restrict__ cent, int S,
                const float* __restrict__ W1, const float* __restrict__ W2, int CO,
                const float* __restrict__ g1, const float* __restrict__ b1,
                const float* __restrict__ m1, const float* __restrict__ v1,
                const float* __restrict__ g2, const float* __restrict__ b2,
                const float* __restrict__ m2, const float* __restrict__ v2,
                float* __restrict__ OUT) {
  extern __shared__ char smem[];
  int ci = 2 * D;
  __bf16* G  = (__bf16*)smem;                 // [32][ci]
  __bf16* H1 = G + (long)ci * 32;             // [32][CO]
  float*  H2 = (float*)(H1 + (long)CO * 32);  // [32][CO]

  int b = blockIdx.x / S, s = blockIdx.x % S;
  const float* Fb = F + (long)b * D * Np;
  const int*   nb = knn + ((long)b * S + s) * 32;
  int cidx = cent[(long)b * S + s];

  for (int i = threadIdx.x; i < ci * 32; i += 256) {
    int j = i & 31, r = i >> 5;
    int nj = nb[j];
    float v;
    if (r < D) v = Fb[(long)r * Np + nj] - Fb[(long)r * Np + cidx];
    else       v = Fb[(long)(r - D) * Np + cidx];
    G[(long)j * ci + r] = (__bf16)v;
  }
  __syncthreads();

  int wave = threadIdx.x >> 5, lane = threadIdx.x & 31;
  int col = lane & 15, hi = lane >> 4;
  int mo  = hi ? 8 : 0;

  // GEMM1: H1 = relu(bn1(W1 * G))
  for (int mt = wave; mt < (CO >> 4); mt += 8) {
    v8f acc[2] = {};
    for (int kb = 0; kb < ci; kb += 32) {
      v16bf a = load_A_fast(W1, ci, mt * 16, kb);
#pragma unroll
      for (int nt = 0; nt < 2; ++nt) {
        const v16bf* bp =
            (const v16bf*)&G[(long)(nt * 16 + col) * ci + kb + (hi ? 16 : 0)];
        acc[nt] = wmma_bf16(a, *bp, acc[nt]);
      }
    }
#pragma unroll
    for (int nt = 0; nt < 2; ++nt) {
      int j = nt * 16 + col;
      v8bf pk;
#pragma unroll
      for (int r = 0; r < 8; ++r) {
        int m = mt * 16 + mo + r;
        float v = acc[nt][r];
        float sc = g1[m] * rsqrtf(v1[m] + 1e-5f);
        v = (v - m1[m]) * sc + b1[m];
        pk[r] = (__bf16)(v > 0.f ? v : 0.f);
      }
      *(v8bf*)&H1[(long)j * CO + mt * 16 + mo] = pk;   // 16B contiguous
    }
  }
  __syncthreads();

  // GEMM2: H2 = relu(bn2(W2 * H1))
  for (int mt = wave; mt < (CO >> 4); mt += 8) {
    v8f acc[2] = {};
    for (int kb = 0; kb < CO; kb += 32) {
      v16bf a = load_A_fast(W2, CO, mt * 16, kb);
#pragma unroll
      for (int nt = 0; nt < 2; ++nt) {
        const v16bf* bp =
            (const v16bf*)&H1[(long)(nt * 16 + col) * CO + kb + (hi ? 16 : 0)];
        acc[nt] = wmma_bf16(a, *bp, acc[nt]);
      }
    }
#pragma unroll
    for (int nt = 0; nt < 2; ++nt) {
      int j = nt * 16 + col;
      float4 f0, f1;
      float* fo[8] = {&f0.x, &f0.y, &f0.z, &f0.w, &f1.x, &f1.y, &f1.z, &f1.w};
#pragma unroll
      for (int r = 0; r < 8; ++r) {
        int m = mt * 16 + mo + r;
        float v = acc[nt][r];
        float sc = g2[m] * rsqrtf(v2[m] + 1e-5f);
        v = (v - m2[m]) * sc + b2[m];
        *fo[r] = v > 0.f ? v : 0.f;
      }
      *(float4*)&H2[(long)j * CO + mt * 16 + mo]     = f0;  // 32B contiguous
      *(float4*)&H2[(long)j * CO + mt * 16 + mo + 4] = f1;
    }
  }
  __syncthreads();

  for (int r = threadIdx.x; r < CO; r += 256) {   // adaptive_max_pool1d(.,1)
    float mx = -1e30f;
    for (int j = 0; j < 32; ++j) mx = fmaxf(mx, H2[(long)j * CO + r]);
    OUT[((long)b * CO + r) * S + s] = mx;
  }
}

// ---------------------------------------------------------------------------
// Farthest point sampling (sequential scan, one block per batch).
// Matches jax: emits the *incoming* `far` each step (first index is 0).
// ---------------------------------------------------------------------------
__global__ __launch_bounds__(256)
void k_fps(const float* __restrict__ xyz, int N, int npoint, int* __restrict__ out) {
  int b = blockIdx.x;
  const float* P = xyz + (long)b * N * 3;
  __shared__ float sval[256];
  __shared__ int   sidx[256];
  __shared__ float cx, cy, cz;
  float dist[8];
  const int PT = (N + 255) / 256;
  for (int i = 0; i < PT; ++i) dist[i] = 1e10f;
  int far = 0;
  for (int it = 0; it < npoint; ++it) {
    if (threadIdx.x == 0) {
      out[(long)b * npoint + it] = far;
      cx = P[(long)far * 3]; cy = P[(long)far * 3 + 1]; cz = P[(long)far * 3 + 2];
    }
    __syncthreads();
    float bv = -1e30f; int bi = 0x7fffffff;
    for (int i = 0; i < PT; ++i) {
      int n = threadIdx.x + i * 256;
      if (n < N) {
        float dx = P[(long)n * 3] - cx, dy = P[(long)n * 3 + 1] - cy, dz = P[(long)n * 3 + 2] - cz;
        float d = dx * dx + dy * dy + dz * dz;
        dist[i] = fminf(dist[i], d);
        if (dist[i] > bv || (dist[i] == bv && n < bi)) { bv = dist[i]; bi = n; }
      }
    }
    sval[threadIdx.x] = bv; sidx[threadIdx.x] = bi;
    __syncthreads();
    for (int st = 128; st > 0; st >>= 1) {
      if (threadIdx.x < st) {
        float ov = sval[threadIdx.x + st]; int oi = sidx[threadIdx.x + st];
        if (ov > sval[threadIdx.x] || (ov == sval[threadIdx.x] && oi < sidx[threadIdx.x])) {
          sval[threadIdx.x] = ov; sidx[threadIdx.x] = oi;
        }
      }
      __syncthreads();
    }
    far = sidx[0];
    __syncthreads();
  }
}

// KNN (k=32): block per (b,s); distances in LDS, 32x argmin extraction.
__global__ __launch_bounds__(256)
void k_knn(const float* __restrict__ xyz, int N,
           const float* __restrict__ q, int S, int* __restrict__ out) {
  extern __shared__ float sd[];
  __shared__ float rv[256];
  __shared__ int   ri[256];
  int b = blockIdx.x / S, s = blockIdx.x % S;
  const float* P = xyz + (long)b * N * 3;
  float qx = q[((long)b * S + s) * 3];
  float qy = q[((long)b * S + s) * 3 + 1];
  float qz = q[((long)b * S + s) * 3 + 2];
  for (int n = threadIdx.x; n < N; n += 256) {
    float dx = P[(long)n * 3] - qx, dy = P[(long)n * 3 + 1] - qy, dz = P[(long)n * 3 + 2] - qz;
    sd[n] = dx * dx + dy * dy + dz * dz;
  }
  __syncthreads();
  for (int kk = 0; kk < 32; ++kk) {
    float bv = 1e30f; int bi = 0x7fffffff;
    for (int n = threadIdx.x; n < N; n += 256)
      if (sd[n] < bv || (sd[n] == bv && n < bi)) { bv = sd[n]; bi = n; }
    rv[threadIdx.x] = bv; ri[threadIdx.x] = bi;
    __syncthreads();
    for (int st = 128; st > 0; st >>= 1) {
      if (threadIdx.x < st) {
        float ov = rv[threadIdx.x + st]; int oi = ri[threadIdx.x + st];
        if (ov < rv[threadIdx.x] || (ov == rv[threadIdx.x] && oi < ri[threadIdx.x])) {
          rv[threadIdx.x] = ov; ri[threadIdx.x] = oi;
        }
      }
      __syncthreads();
    }
    if (threadIdx.x == 0) {
      out[((long)b * S + s) * 32 + kk] = ri[0];
      sd[ri[0]] = 1e30f;
    }
    __syncthreads();
  }
}

// ---------------------------------------------------------------------------
// Small elementwise / reduction kernels
// ---------------------------------------------------------------------------
__global__ void k_txyz(const float* __restrict__ x, float* __restrict__ xc, long tot, int Nn) {
  long i = (long)blockIdx.x * 256 + threadIdx.x;
  if (i >= tot) return;
  int n = i % Nn; long r = i / Nn; int k = r % 3; long b = r / 3;
  xc[i] = x[(b * Nn + n) * 3 + k];
}

__global__ void k_gather3(const float* __restrict__ xyz, int N,
                          const int* __restrict__ idx, int S,
                          float* __restrict__ out, long tot) {
  long i = (long)blockIdx.x * 256 + threadIdx.x;
  if (i >= tot) return;
  int k = i % 3; long r = i / 3; int s = r % S; long b = r / S;
  out[i] = xyz[(b * N + idx[b * S + s]) * 3 + k];
}

__global__ void k_sub(const float* __restrict__ a, const float* __restrict__ c,
                      float* __restrict__ y, long tot) {
  long i = (long)blockIdx.x * 256 + threadIdx.x;
  if (i < tot) y[i] = a[i] - c[i];
}

// x += t; also mirror into concat buffer at row offset (pt_last outs list).
__global__ void k_addcopy(float* __restrict__ x, const float* __restrict__ t,
                          float* __restrict__ cc, int rowoff, int C, int Nn,
                          int ccRows, long tot) {
  long i = (long)blockIdx.x * 256 + threadIdx.x;
  if (i >= tot) return;
  int n = i % Nn; long r = i / Nn; int c = r % C; long b = r / C;
  float v = x[i] + t[i];
  x[i] = v;
  cc[(b * ccRows + rowoff + c) * (long)Nn + n] = v;
}

__global__ void k_copyrows(const float* __restrict__ src, float* __restrict__ cc,
                           int rowoff, int C, int Nn, int ccRows, long tot) {
  long i = (long)blockIdx.x * 256 + threadIdx.x;
  if (i >= tot) return;
  int n = i % Nn; long r = i / Nn; int c = r % C; long b = r / C;
  cc[(b * ccRows + rowoff + c) * (long)Nn + n] = src[i];
}

// softmax over last dim (row length == Nn <= 256), in place
__global__ __launch_bounds__(256)
void k_softmax(float* __restrict__ E, int Nn) {
  __shared__ float red[256];
  float* p = E + (long)blockIdx.x * Nn;
  int t = threadIdx.x;
  float v = (t < Nn) ? p[t] : -1e30f;
  red[t] = v; __syncthreads();
  for (int st = 128; st > 0; st >>= 1) {
    if (t < st) red[t] = fmaxf(red[t], red[t + st]);
    __syncthreads();
  }
  float mx = red[0]; __syncthreads();
  float e = (t < Nn) ? __expf(v - mx) : 0.f;
  red[t] = e; __syncthreads();
  for (int st = 128; st > 0; st >>= 1) {
    if (t < st) red[t] += red[t + st];
    __syncthreads();
  }
  if (t < Nn) p[t] = e / red[0];
}

// att /= (1e-9 + att.sum(dim=1)) : column renorm, block per (b, column m)
__global__ __launch_bounds__(256)
void k_attnorm(float* __restrict__ A, int Nn) {
  __shared__ float red[256];
  int b = blockIdx.x / Nn, m = blockIdx.x % Nn;
  float* base = A + (long)b * Nn * Nn;
  int t = threadIdx.x;
  float v = (t < Nn) ? base[(long)t * Nn + m] : 0.f;
  red[t] = v; __syncthreads();
  for (int st = 128; st > 0; st >>= 1) {
    if (t < st) red[t] += red[t + st];
    __syncthreads();
  }
  float s = red[0] + 1e-9f;
  if (t < Nn) base[(long)t * Nn + m] = v / s;
}

// per-channel max & mean over N -> gf[b][0:C]=max, gf[b][C:2C]=mean
__global__ __launch_bounds__(256)
void k_pool(const float* __restrict__ X, int C, int Nn, float* __restrict__ gf) {
  __shared__ float rmx[256], rsm[256];
  int c = blockIdx.x, b = blockIdx.y, t = threadIdx.x;
  const float* p = X + ((long)b * C + c) * Nn;
  float mx = -1e30f, sm = 0.f;
  for (int n = t; n < Nn; n += 256) { float v = p[n]; mx = fmaxf(mx, v); sm += v; }
  rmx[t] = mx; rsm[t] = sm; __syncthreads();
  for (int st = 128; st > 0; st >>= 1) {
    if (t < st) { rmx[t] = fmaxf(rmx[t], rmx[t + st]); rsm[t] += rsm[t + st]; }
    __syncthreads();
  }
  if (t == 0) {
    gf[(long)b * 2 * C + c]     = rmx[0];
    gf[(long)b * 2 * C + C + c] = rsm[0] / (float)Nn;
  }
}

// y0[b][o] = dot(Wc[o][0:Kk], gf[b][0:Kk])  (rank-1 global-feature term)
__global__ __launch_bounds__(256)
void k_matvec(const float* __restrict__ Wc, long ld, const float* __restrict__ gf,
              int Kk, float* __restrict__ y0, int M) {
  __shared__ float red[256];
  int o = blockIdx.x, b = blockIdx.y, t = threadIdx.x;
  float s = 0.f;
  for (int k = t; k < Kk; k += 256) s += Wc[(long)o * ld + k] * gf[(long)b * Kk + k];
  red[t] = s; __syncthreads();
  for (int st = 128; st > 0; st >>= 1) {
    if (t < st) red[t] += red[t + st];
    __syncthreads();
  }
  if (t == 0) y0[(long)b * M + o] = red[0];
}

// ---------------------------------------------------------------------------
// Host orchestration
// ---------------------------------------------------------------------------
extern "C" void kernel_launch(void* const* d_in, const int* in_sizes, int n_in,
                              void* d_out, int out_size, void* d_ws, size_t ws_size,
                              hipStream_t stream) {
  const int B = 16, N = 2048, S0 = 512, S1 = 256;

  // --- inputs, jax pytree-leaf order (dict keys sorted) ---
  const float* x = (const float*)d_in[0];
  int pi = 1;
  auto F = [&]() { return (const float*)d_in[pi++]; };
  const float *bn1_b=F(),*bn1_g=F(),*bn1_m=F(),*bn1_v=F();
  const float *bn2_b=F(),*bn2_g=F(),*bn2_m=F(),*bn2_v=F();
  const float *bns1_b=F(),*bns1_g=F(),*bns1_m=F(),*bns1_v=F();
  const float *bns2_b=F(),*bns2_g=F(),*bns2_m=F(),*bns2_v=F();
  const float *bns3_b=F(),*bns3_g=F(),*bns3_m=F(),*bns3_v=F();
  const float *c1_b=F(),*c1_w=F(),*c2_b=F(),*c2_w=F(),*c3_b=F(),*c3_w=F();
  const float *c4_b=F(),*c4_w=F();
  const float *conv1_w=F(),*conv2_w=F();
  const float *fbn_b=F(),*fbn_g=F(),*fbn_m=F(),*fbn_v=F();
  const float *fuse_w=F();
  const float *g0b1_b=F(),*g0b1_g=F(),*g0b1_m=F(),*g0b1_v=F();
  const float *g0b2_b=F(),*g0b2_g=F(),*g0b2_m=F(),*g0b2_v=F();
  const float *g0_w1=F(),*g0_w2=F();
  const float *g1b1_b=F(),*g1b1_g=F(),*g1b1_m=F(),*g1b1_v=F();
  const float *g1b2_b=F(),*g1b2_g=F(),*g1b2_m=F(),*g1b2_v=F();
  const float *g1_w1=F(),*g1_w2=F();
  const float *pb1_b=F(),*pb1_g=F(),*pb1_m=F(),*pb1_v=F();
  const float *pb2_b=F(),*pb2_g=F(),*pb2_m=F(),*pb2_v=F();
  const float *sa_qk[4],*sa_tb[4],*sa_tw[4],*sa_vb[4],*sa_vw[4];
  for (int l = 0; l < 4; ++l) { sa_qk[l]=F(); sa_tb[l]=F(); sa_tw[l]=F(); sa_vb[l]=F(); sa_vw[l]=F(); }
  const float *ptl_w1=F(),*ptl_w2=F();

  // --- workspace carve ---
  char* wp = (char*)d_ws;
  auto AF = [&](size_t n) { float* p = (float*)wp; wp += n * sizeof(float); return p; };
  auto AI = [&](size_t n) { int*   p = (int*)wp;   wp += n * sizeof(int);   return p; };
  float* xc    = AF((size_t)B*3*N);
  float* t128  = AF((size_t)B*128*N);
  float* PF    = AF((size_t)B*128*N);        // points_feature
  int*   fi0   = AI((size_t)B*S0);
  float* nxyz0 = AF((size_t)B*S0*3);
  int*   knn0  = AI((size_t)B*S0*32);
  float* F0    = AF((size_t)B*256*S0);       // feature_0
  int*   fi1   = AI((size_t)B*S1);
  float* nxyz1 = AF((size_t)B*S1*3);
  int*   knn1  = AI((size_t)B*S1*32);
  float* F1    = AF((size_t)B*512*S1);       // feature_1
  float* ptA   = AF((size_t)B*512*S1);
  float* X512  = AF((size_t)B*512*S1);       // running x in pt_last
  float* QK    = AF((size_t)B*128*S1);
  float* XV    = AF((size_t)B*512*S1);
  float* EN    = AF((size_t)B*S1*S1);
  float* XR    = AF((size_t)B*512*S1);
  float* U     = AF((size_t)B*512*S1);
  float* TB    = AF((size_t)B*512*S1);
  float* CC    = AF((size_t)B*2560*S1);      // concat [2048 sa outs | 512 f1]
  float* FU    = AF((size_t)B*1024*S1);      // fused
  float* GV    = AF((size_t)B*2048);         // [max | mean]
  float* Y0    = AF((size_t)B*1024);         // rank-1 c1 term
  float* H1    = AF((size_t)B*1024*N);
  float* H2    = AF((size_t)B*512*N);
  float* H3    = AF((size_t)B*256*N);
  (void)ws_size; (void)n_in; (void)in_sizes; (void)out_size;

  auto gemm = [&](const float* W, long wbs, long ld, int tr,
                  const float* Xp, long xbs, float* Y, long ybs,
                  int M, int K, int Nn,
                  const float* bias, const float* bias2,
                  const float* g, const float* bb, const float* mm, const float* vv,
                  int act) {
    dim3 grid((Nn + 63) / 64, (M + 127) / 128, B);
    k_gemm<<<grid, 256, 0, stream>>>(W, wbs, ld, tr, Xp, xbs, Y, ybs,
                                     M, K, Nn, bias, bias2, g, bb, mm, vv, act);
  };
  auto flat = [&](long tot) { return dim3((unsigned)((tot + 255) / 256)); };

  // ---- backbone MLP ----
  k_txyz<<<flat((long)B*3*N), 256, 0, stream>>>(x, xc, (long)B*3*N, N);
  gemm(conv1_w, 0, 3,   0, xc,   (long)3*N,   t128, (long)128*N, 128, 3,   N, 0, 0, bn1_g, bn1_b, bn1_m, bn1_v, 1);
  gemm(conv2_w, 0, 128, 0, t128, (long)128*N, PF,   (long)128*N, 128, 128, N, 0, 0, bn2_g, bn2_b, bn2_m, bn2_v, 1);

  // ---- stage 0: sample_and_group(512,32) + local_op gl0 ----
  k_fps<<<B, 256, 0, stream>>>(x, N, S0, fi0);
  k_gather3<<<flat((long)B*S0*3), 256, 0, stream>>>(x, N, fi0, S0, nxyz0, (long)B*S0*3);
  k_knn<<<B*S0, 256, (size_t)N*4, stream>>>(x, N, nxyz0, S0, knn0);
  {
    size_t sm = (size_t)(256*32 + 256*32)*2 + (size_t)256*32*4;   // 64 KB
    k_local_op<<<B*S0, 256, sm, stream>>>(PF, 128, N, knn0, fi0, S0,
        g0_w1, g0_w2, 256, g0b1_g, g0b1_b, g0b1_m, g0b1_v,
        g0b2_g, g0b2_b, g0b2_m, g0b2_v, F0);
  }

  // ---- stage 1: sample_and_group(256,32) + local_op gl1 ----
  k_fps<<<B, 256, 0, stream>>>(nxyz0, S0, S1, fi1);
  k_gather3<<<flat((long)B*S1*3), 256, 0, stream>>>(nxyz0, S0, fi1, S1, nxyz1, (long)B*S1*3);
  k_knn<<<B*S1, 256, (size_t)S0*4, stream>>>(nxyz0, S0, nxyz1, S1, knn1);
  {
    size_t sm = (size_t)(512*32 + 512*32)*2 + (size_t)512*32*4;   // 128 KB (<320KB/WGP)
    k_local_op<<<B*S1, 256, sm, stream>>>(F0, 256, S0, knn1, fi1, S1,
        g1_w1, g1_w2, 512, g1b1_g, g1b1_b, g1b1_m, g1b1_v,
        g1b2_g, g1b2_b, g1b2_m, g1b2_v, F1);
  }

  // ---- pt_last ----
  long s512 = (long)512*S1, s128 = (long)128*S1, sEE = (long)S1*S1;
  gemm(ptl_w1, 0, 512, 0, F1,  s512, ptA,  s512, 512, 512, S1, 0, 0, pb1_g, pb1_b, pb1_m, pb1_v, 1);
  gemm(ptl_w2, 0, 512, 0, ptA, s512, X512, s512, 512, 512, S1, 0, 0, pb2_g, pb2_b, pb2_m, pb2_v, 1);
  for (int l = 0; l < 4; ++l) {
    gemm(sa_qk[l], 0, 512, 0, X512, s512, QK, s128, 128, 512, S1, 0,        0, 0,0,0,0, 0);
    gemm(sa_vw[l], 0, 512, 0, X512, s512, XV, s512, 512, 512, S1, sa_vb[l], 0, 0,0,0,0, 0);
    // energy = (QK)^T * QK : A batched & transposed, X batched
    gemm(QK, s128, S1, 1, QK, s128, EN, sEE, S1, 128, S1, 0, 0, 0,0,0,0, 0);
    k_softmax<<<B*S1, 256, 0, stream>>>(EN, S1);
    k_attnorm<<<B*S1, 256, 0, stream>>>(EN, S1);
    // xr = XV * att : A batched
    gemm(XV, s512, S1, 0, EN, sEE, XR, s512, 512, S1, S1, 0, 0, 0,0,0,0, 0);
    k_sub<<<flat((long)B*s512), 256, 0, stream>>>(X512, XR, U, (long)B*s512);
    gemm(sa_tw[l], 0, 512, 0, U, s512, TB, s512, 512, 512, S1, sa_tb[l], 0, 0,0,0,0, 1);
    k_addcopy<<<flat((long)B*s512), 256, 0, stream>>>(X512, TB, CC, l*512, 512, S1, 2560, (long)B*s512);
  }
  k_copyrows<<<flat((long)B*s512), 256, 0, stream>>>(F1, CC, 2048, 512, S1, 2560, (long)B*s512);

  // ---- fuse conv + leaky relu ----
  gemm(fuse_w, 0, 2560, 0, CC, (long)2560*S1, FU, (long)1024*S1,
       1024, 2560, S1, 0, 0, fbn_g, fbn_b, fbn_m, fbn_v, 2);

  // ---- global feature (rank-1 along N) ----
  k_pool<<<dim3(1024, B), 256, 0, stream>>>(FU, 1024, S1, GV);
  k_matvec<<<dim3(1024, B), 256, 0, stream>>>(c1_w, 2176, GV, 2048, Y0, 1024);

  // ---- segmentation head at N=2048 ----
  gemm(c1_w + 2048, 0, 2176, 0, PF, (long)128*N, H1, (long)1024*N,
       1024, 128, N, c1_b, Y0, bns1_g, bns1_b, bns1_m, bns1_v, 1);
  gemm(c2_w, 0, 1024, 0, H1, (long)1024*N, H2, (long)512*N,
       512, 1024, N, c2_b, 0, bns2_g, bns2_b, bns2_m, bns2_v, 1);
  gemm(c3_w, 0, 512, 0, H2, (long)512*N, H3, (long)256*N,
       256, 512, N, c3_b, 0, bns3_g, bns3_b, bns3_m, bns3_v, 1);
  gemm(c4_w, 0, 256, 0, H3, (long)256*N, (float*)d_out, (long)7*N,
       7, 256, N, c4_b, 0, 0,0,0,0, 0);
}
